// MultiHeadSelfAttention_15496242004485
// MI455X (gfx1250) — matrix operand order repaired and checked by reading, not compiled, not run on hardware
//
#include <hip/hip_runtime.h>

typedef __attribute__((ext_vector_type(16))) __bf16 v16bf;
typedef __attribute__((ext_vector_type(8)))  float  v8f;

#define B_   4
#define S_   2048
#define D_   1024
#define H_   16
#define DH_  64
#define N3_  3072
#define NEGV (-1000000000.0f)
// ln(10000)/64
#define ROPE_LN 0.14391156831212787f

// ---------------------------------------------------------------------------
// WMMA helper: D(f32 16x16) = A(bf16 16x32) * B(bf16 32x16) + C
// ---------------------------------------------------------------------------
__device__ __forceinline__ v8f wmma_bf16(v16bf a, v16bf b, v8f c) {
  return __builtin_amdgcn_wmma_f32_16x16x32_bf16(
      /*neg_a=*/false, a, /*neg_b=*/false, b,
      /*c_mod=*/(short)0, c, /*reuse_a=*/false, /*reuse_b=*/false);
}

// Async global->LDS b128: per-lane 16B moved without VGPR staging (ASYNCcnt).
__device__ __forceinline__ void async_ld_b128(unsigned lds_off, const void* g) {
  asm volatile("global_load_async_to_lds_b128 %0, %1, off"
               :: "v"(lds_off), "v"(g) : "memory");
}

// A-operand (16x32 bf16).  ISA 7.12.2: lane half hi adds K+=8; element pairs
// live at K = [8hi..8hi+7] and [16+8hi..23+8hi]  -> two contiguous b128 loads.
__device__ __forceinline__ v16bf load_a_frag(const __bf16* p, int mrow, int kcol,
                                             int ld, int lane) {
  int m  = mrow + (lane & 15);
  int hi = lane >> 4;
  const __bf16* base = p + (size_t)m * ld + kcol + 8 * hi;
  union { uint4 q[2]; v16bf v; } u;
  u.q[0] = *(const uint4*)(base);
  u.q[1] = *(const uint4*)(base + 16);
  return u.v;
}

// B-operand (32x16 bf16) from an n-major LDS tile (elements contiguous in K).
// Lane element e -> K = krow+8hi+e (e<8), krow+16+8hi+(e-8) (e>=8), N = nbase + lane&15 -> two b128 loads.
__device__ __forceinline__ v16bf load_b_frag(const __bf16* p, int nbase, int ld,
                                             int krow, int lane) {
  int n  = nbase + (lane & 15);
  int hi = lane >> 4;
  const __bf16* base = p + (size_t)n * ld + krow + 8 * hi;
  union { uint4 q[2]; v16bf v; } u;
  u.q[0] = *(const uint4*)(base);
  u.q[1] = *(const uint4*)(base + 16);
  return u.v;
}

// ---------------------------------------------------------------------------
// fp32 -> bf16 conversion (x), vectorized by 4
// ---------------------------------------------------------------------------
__global__ void __launch_bounds__(256) cvt_f32_bf16(const float* __restrict__ s,
                                                    __bf16* __restrict__ d, int n4) {
  int i = blockIdx.x * blockDim.x + threadIdx.x;
  if (i < n4) {
    float4 f = ((const float4*)s)[i];
    d[4 * i + 0] = (__bf16)f.x;
    d[4 * i + 1] = (__bf16)f.y;
    d[4 * i + 2] = (__bf16)f.z;
    d[4 * i + 3] = (__bf16)f.w;
  }
}

// ---------------------------------------------------------------------------
// fp32 [K][N] -> bf16 [N][K] convert+transpose, 64x64 LDS tiles.
// ---------------------------------------------------------------------------
__global__ void __launch_bounds__(256) cvt_transpose(const float* __restrict__ src,
                                                     __bf16* __restrict__ dst,
                                                     int K, int N) {
  __shared__ __bf16 tile[64][72];   // +8 pad
  const int tid = threadIdx.x;
  const int tn0 = blockIdx.x * 64, tk0 = blockIdx.y * 64;
#pragma unroll
  for (int t = 0; t < 4; ++t) {
    int vi = tid + t * 256;               // 1024 float4
    int r = vi >> 4, c = (vi & 15) * 4;
    float4 f = *(const float4*)(src + (size_t)(tk0 + r) * N + tn0 + c);
    tile[c + 0][r] = (__bf16)f.x;
    tile[c + 1][r] = (__bf16)f.y;
    tile[c + 2][r] = (__bf16)f.z;
    tile[c + 3][r] = (__bf16)f.w;
  }
  __syncthreads();
#pragma unroll
  for (int t = 0; t < 2; ++t) {
    int vi = tid + t * 256;               // 512 uint4
    int n = vi >> 3, kc = (vi & 7) * 8;
    *(uint4*)(dst + (size_t)(tn0 + n) * K + tk0 + kc) = *(const uint4*)&tile[n][kc];
  }
}

// ---------------------------------------------------------------------------
// GEMM tile issue: async-copy one 128x32 A tile and one 128x32 B tile.
// 4 async b128 instructions per wave per call (ASYNCcnt += 4).
// ---------------------------------------------------------------------------
__device__ __forceinline__ void gemm_issue(const __bf16* A, const __bf16* WT,
                                           int rowBase, int colBase, int k0,
                                           unsigned offA, unsigned offB, int tid) {
#pragma unroll
  for (int t = 0; t < 2; ++t) {
    int vi = tid + t * 256;
    int r = vi >> 2, c = (vi & 3) * 8;
    async_ld_b128(offA + vi * 16, A  + (size_t)(rowBase + r) * D_ + k0 + c);
    async_ld_b128(offB + vi * 16, WT + (size_t)(colBase + r) * D_ + k0 + c);
  }
}

__device__ __forceinline__ void gemm_compute(const char* smem, int cur,
                                             int wm, int wn, int lane,
                                             v8f (&acc)[4][2]) {
  const __bf16* ldsA = (const __bf16*)(smem + cur * 8192);
  const __bf16* ldsB = (const __bf16*)(smem + 16384 + cur * 8192);
  v16bf bf0 = load_b_frag(ldsB, wn * 32,      32, 0, lane);
  v16bf bf1 = load_b_frag(ldsB, wn * 32 + 16, 32, 0, lane);
#pragma unroll
  for (int i = 0; i < 4; ++i) {
    v16bf af = load_a_frag(ldsA, wm * 64 + i * 16, 0, 32, lane);
    acc[i][0] = wmma_bf16(af, bf0, acc[i][0]);
    acc[i][1] = wmma_bf16(af, bf1, acc[i][1]);
  }
}

// ---------------------------------------------------------------------------
// Shared GEMM mainloop: branch-free body, drain iteration hoisted out.
// LDS: A0@0 A1@8K B0@16K B1@24K.
// ---------------------------------------------------------------------------
__device__ __forceinline__ void gemm_main(const __bf16* __restrict__ A,
                                          const __bf16* __restrict__ WT,
                                          int rowBase, int colBase,
                                          char* smem, v8f (&acc)[4][2]) {
  const int tid  = threadIdx.x;
  const int lane = tid & 31;
  const int wave = tid >> 5;
  const int wm   = wave >> 2;   // 0..1 (64 rows)
  const int wn   = wave & 3;    // 0..3 (32 cols)
  const unsigned ldsBase = __builtin_amdgcn_groupstaticsize();

  gemm_issue(A, WT, rowBase, colBase, 0, ldsBase, ldsBase + 16384, tid);

  for (int it = 0; it < 31; ++it) {
    const int cur = it & 1, nxt = cur ^ 1;
    gemm_issue(A, WT, rowBase, colBase, (it + 1) * 32,
               ldsBase + nxt * 8192, ldsBase + 16384 + nxt * 8192, tid);
    asm volatile("s_wait_asynccnt 4" ::: "memory");   // tile `it` resident
    __syncthreads();
    gemm_compute(smem, cur, wm, wn, lane, acc);
    __syncthreads();   // all waves done with buffer `cur` before re-issue
  }
  // drain: tile 31 lives in buffer 1
  asm volatile("s_wait_asynccnt 0" ::: "memory");
  __syncthreads();
  gemm_compute(smem, 1, wm, wn, lane, acc);
}

// ---------------------------------------------------------------------------
// GEMM1: qkv = x(8192x1024) @ Wqkv, RoPE epilogue, scatter to
//   q  [b][h][s][d], k [b][h][s][d], vT [b][h][d][s]   (all bf16)
// LDS: A/B double buffers 32KB | C 128x128 f32 64KB  (96KB dynamic)
// ---------------------------------------------------------------------------
__global__ void __launch_bounds__(256)
__attribute__((amdgpu_waves_per_eu(1)))
gemm_qkv_rope(const __bf16* __restrict__ A, const __bf16* __restrict__ WT,
              __bf16* __restrict__ qb, __bf16* __restrict__ kb,
              __bf16* __restrict__ vtb) {
  extern __shared__ char smem[];
  float* ldsC = (float*)(smem + 32768);

  const int tid  = threadIdx.x;
  const int lane = tid & 31;
  const int wave = tid >> 5;
  const int wm = wave >> 2, wn = wave & 3;
  const int rowBase = blockIdx.x * 128;
  const int colBase = blockIdx.y * 128;

  v8f acc[4][2] = {};
  gemm_main(A, WT, rowBase, colBase, smem, acc);

  {
    const int hi = lane >> 4, nl = lane & 15;
#pragma unroll
    for (int i = 0; i < 4; ++i)
#pragma unroll
      for (int j = 0; j < 2; ++j)
#pragma unroll
        for (int r = 0; r < 8; ++r)
          ldsC[(wm * 64 + i * 16 + r + 8 * hi) * 128 + wn * 32 + j * 16 + nl] =
              acc[i][j][r];
  }
  __syncthreads();

  for (int idx = tid; idx < 128 * 64; idx += 256) {
    int row = idx >> 6;
    int c0  = (idx & 63) * 2;
    float x1 = ldsC[row * 128 + c0];
    float x2 = ldsC[row * 128 + c0 + 1];
    int gm = rowBase + row;
    int b  = gm >> 11;
    int s  = gm & 2047;
    int gc = colBase + c0;
    int part = gc >> 10;         // 0=q 1=k 2=v
    int w    = gc & 1023;
    int h    = w >> 6;
    int d    = w & 63;           // even
    if (part == 2) {
      __bf16* vp = vtb + ((size_t)(b * H_ + h) * DH_ + d) * S_ + s;
      vp[0]  = (__bf16)x1;
      vp[S_] = (__bf16)x2;       // row d+1, same column s
    } else {
      float theta = __expf(-(float)d * ROPE_LN);   // 10000^(-d/64)
      float ang   = (float)s * theta;
      float sn, cs;
      __sincosf(ang, &sn, &cs);
      float r1 = x1 * cs - x2 * sn;
      float r2 = x1 * sn + x2 * cs;
      __bf16* tp = (part == 0 ? qb : kb) +
                   (((size_t)(b * H_ + h) * S_ + s) * DH_ + d);
      tp[0] = (__bf16)r1;
      tp[1] = (__bf16)r2;
    }
  }
}

// ---------------------------------------------------------------------------
// Attention helpers
// ---------------------------------------------------------------------------
__device__ __forceinline__ void attn_issue(const __bf16* kbase, const __bf16* vtbase,
                                           int s0, unsigned offK, unsigned offV,
                                           int tid) {
#pragma unroll
  for (int t = 0; t < 4; ++t) {
    int vi = tid + t * 128;
    int r = vi >> 3, c = (vi & 7) * 8;
    async_ld_b128(offK + vi * 16, kbase  + (size_t)(s0 + r) * DH_ + c);
    async_ld_b128(offV + vi * 16, vtbase + (size_t)r * S_ + s0 + c);
  }
}

// One 64-key flash-attention step for one wave (16 query rows).
__device__ __forceinline__ void attn_tile(const char* smem, __bf16* ldsP,
                                          int j, int cur, int qrow0, int wave,
                                          int lane, int hi, int nl,
                                          const v16bf (&qa)[2], v8f (&o)[4],
                                          float (&mrow)[8], float (&lrow)[8]) {
  const __bf16* ldsK  = (const __bf16*)(smem + cur * 8192);
  const __bf16* ldsVt = (const __bf16*)(smem + 16384 + cur * 8192);
  const float scale = 0.125f;

  // Scores S = Q K^T  (16x64 per wave).
  float sv[4][8];
#pragma unroll
  for (int kn = 0; kn < 4; ++kn) {
    v8f c = {};
    c = wmma_bf16(qa[0], load_b_frag(ldsK, kn * 16, 64, 0,  lane), c);
    c = wmma_bf16(qa[1], load_b_frag(ldsK, kn * 16, 64, 32, lane), c);
#pragma unroll
    for (int r = 0; r < 8; ++r) {
      int qi = qrow0 + r + 8 * hi;
      int kj = j * 64 + kn * 16 + nl;
      sv[kn][r] = c[r] * scale + ((qi == kj) ? NEGV : 0.f);
    }
  }

  // Online softmax (rows live in 16-lane halves under wave32).
#pragma unroll
  for (int r = 0; r < 8; ++r) {
    float mb = fmaxf(fmaxf(sv[0][r], sv[1][r]), fmaxf(sv[2][r], sv[3][r]));
#pragma unroll
    for (int off = 1; off < 16; off <<= 1)
      mb = fmaxf(mb, __shfl_xor(mb, off, 32));
    float mn    = fmaxf(mrow[r], mb);
    float alpha = __expf(mrow[r] - mn);
    mrow[r] = mn;
    float rs = 0.f;
#pragma unroll
    for (int kn = 0; kn < 4; ++kn) {
      float p = __expf(sv[kn][r] - mn);
      sv[kn][r] = p;
      rs += p;
    }
#pragma unroll
    for (int off = 1; off < 16; off <<= 1) rs += __shfl_xor(rs, off, 32);
    lrow[r] = lrow[r] * alpha + rs;
#pragma unroll
    for (int dn = 0; dn < 4; ++dn) o[dn][r] = o[dn][r] * alpha;
  }

  // Restage P (C layout) -> A layout via per-wave LDS region.
  __bf16* pW = ldsP + wave * 16 * 64;
#pragma unroll
  for (int kn = 0; kn < 4; ++kn)
#pragma unroll
    for (int r = 0; r < 8; ++r)
      pW[(r + 8 * hi) * 64 + kn * 16 + nl] = (__bf16)sv[kn][r];

  v16bf pa0 = load_a_frag(pW, 0, 0,  64, lane);
  v16bf pa1 = load_a_frag(pW, 0, 32, 64, lane);

  // O += P V
#pragma unroll
  for (int dn = 0; dn < 4; ++dn) {
    o[dn] = wmma_bf16(pa0, load_b_frag(ldsVt, dn * 16, 64, 0,  lane), o[dn]);
    o[dn] = wmma_bf16(pa1, load_b_frag(ldsVt, dn * 16, 64, 32, lane), o[dn]);
  }
}

// ---------------------------------------------------------------------------
// Flash attention: grid (S/64, B*H), 128 threads (4 waves x 16 query rows).
// K natural [s][d] -> LDS [key][d] ; V transposed [d][s] -> LDS [d][key].
// Double-buffered async tiles: K0@0 K1@8K V0@16K V1@24K | P@32K  (40KB)
// ---------------------------------------------------------------------------
__global__ void __launch_bounds__(128)
__attribute__((amdgpu_waves_per_eu(1)))
attn_fwd(const __bf16* __restrict__ q, const __bf16* __restrict__ k,
         const __bf16* __restrict__ vT, __bf16* __restrict__ y) {
  extern __shared__ char smem[];
  __bf16* ldsP = (__bf16*)(smem + 32768);       // per-wave 16x64

  const int tid  = threadIdx.x;
  const int lane = tid & 31;
  const int wave = tid >> 5;
  const int bh   = blockIdx.y;
  const int qrow0 = blockIdx.x * 64 + wave * 16;
  const int hi = lane >> 4, nl = lane & 15;
  const unsigned ldsBase = __builtin_amdgcn_groupstaticsize();

  const __bf16* qbase  = q  + (size_t)bh * S_ * DH_;
  const __bf16* kbase  = k  + (size_t)bh * S_ * DH_;
  const __bf16* vtbase = vT + (size_t)bh * DH_ * S_;

  v16bf qa[2];
  qa[0] = load_a_frag(qbase, qrow0, 0,  DH_, lane);
  qa[1] = load_a_frag(qbase, qrow0, 32, DH_, lane);

  v8f o[4] = {};
  float mrow[8], lrow[8];
#pragma unroll
  for (int r = 0; r < 8; ++r) { mrow[r] = -1e30f; lrow[r] = 0.f; }

  attn_issue(kbase, vtbase, 0, ldsBase, ldsBase + 16384, tid);

  for (int j = 0; j < 31; ++j) {
    const int cur = j & 1, nxt = cur ^ 1;
    attn_issue(kbase, vtbase, (j + 1) * 64,
               ldsBase + nxt * 8192, ldsBase + 16384 + nxt * 8192, tid);
    asm volatile("s_wait_asynccnt 8" ::: "memory");   // tile j resident
    __syncthreads();
    attn_tile(smem, ldsP, j, cur, qrow0, wave, lane, hi, nl, qa, o, mrow, lrow);
    __syncthreads();   // all waves done with buffer `cur` before re-issue
  }
  // drain: tile 31 lives in buffer 1
  asm volatile("s_wait_asynccnt 0" ::: "memory");
  __syncthreads();
  attn_tile(smem, ldsP, 31, 1, qrow0, wave, lane, hi, nl, qa, o, mrow, lrow);

  const int b = bh >> 4, h = bh & 15;
#pragma unroll
  for (int r = 0; r < 8; ++r) {
    float inv = 1.f / lrow[r];
    int qi = qrow0 + r + 8 * hi;
#pragma unroll
    for (int dn = 0; dn < 4; ++dn)
      y[((size_t)(b * S_ + qi)) * D_ + h * DH_ + dn * 16 + nl] =
          (__bf16)(o[dn][r] * inv);
  }
}

// ---------------------------------------------------------------------------
// GEMM2: out = y(8192x1024 bf16) @ Wproj -> fp32 d_out.  LDS 32KB.
// ---------------------------------------------------------------------------
__global__ void __launch_bounds__(256)
__attribute__((amdgpu_waves_per_eu(1)))
gemm_proj(const __bf16* __restrict__ A, const __bf16* __restrict__ WT,
          float* __restrict__ out) {
  extern __shared__ char smem[];

  const int tid  = threadIdx.x;
  const int lane = tid & 31;
  const int wave = tid >> 5;
  const int wm = wave >> 2, wn = wave & 3;
  const int rowBase = blockIdx.x * 128;
  const int colBase = blockIdx.y * 128;
  const int hi = lane >> 4, nl = lane & 15;

  v8f acc[4][2] = {};
  gemm_main(A, WT, rowBase, colBase, smem, acc);

#pragma unroll
  for (int i = 0; i < 4; ++i)
#pragma unroll
    for (int j = 0; j < 2; ++j)
#pragma unroll
      for (int r = 0; r < 8; ++r) {
        int row = rowBase + wm * 64 + i * 16 + r + 8 * hi;
        int col = colBase + wn * 32 + j * 16 + nl;
        out[(size_t)row * D_ + col] = acc[i][j][r];
      }
}

// ---------------------------------------------------------------------------
// Host launcher
// ---------------------------------------------------------------------------
extern "C" void kernel_launch(void* const* d_in, const int* in_sizes, int n_in,
                              void* d_out, int out_size, void* d_ws, size_t ws_size,
                              hipStream_t stream) {
  const float* x     = (const float*)d_in[0];
  const float* Wqkv  = (const float*)d_in[1];
  const float* Wproj = (const float*)d_in[2];
  float* out = (float*)d_out;

  char* ws = (char*)d_ws;
  __bf16* xb      = (__bf16*)(ws + 0);          // 16 MB  x bf16
  __bf16* wqkvT   = (__bf16*)(ws + 16777216);   //  6 MB  Wqkv^T bf16 [3072][1024]
  __bf16* wprojT  = (__bf16*)(ws + 23068672);   //  2 MB  Wproj^T bf16 [1024][1024]
  __bf16* qbuf    = (__bf16*)(ws + 25165824);   // 16 MB  q  [b][h][s][d]
  __bf16* kbuf    = (__bf16*)(ws + 41943040);   // 16 MB  k  [b][h][s][d]
  __bf16* vtbuf   = (__bf16*)(ws + 58720256);   // 16 MB  vT [b][h][d][s]
  __bf16* ybuf    = (__bf16*)(ws + 75497472);   // 16 MB  y  [b][s][h*d]

  {
    int n4 = (B_ * S_ * D_) / 4;
    cvt_f32_bf16<<<(n4 + 255) / 256, 256, 0, stream>>>(x, xb, n4);
  }
  cvt_transpose<<<dim3(N3_ / 64, D_ / 64), 256, 0, stream>>>(Wqkv,  wqkvT,  D_, N3_);
  cvt_transpose<<<dim3(D_ / 64,  D_ / 64), 256, 0, stream>>>(Wproj, wprojT, D_, D_);

  gemm_qkv_rope<<<dim3(64, 24), 256, 98304, stream>>>(xb, wqkvT, qbuf, kbuf, vtbuf);
  attn_fwd<<<dim3(32, 64), 128, 40960, stream>>>(qbuf, kbuf, vtbuf, ybuf);
  gemm_proj<<<dim3(64, 8), 256, 32768, stream>>>(ybuf, wprojT, out);
}
